// MultimodalProjector_43250320671027
// MI455X (gfx1250) — matrix operand order repaired and checked by reading, not compiled, not run on hardware
//
#include <hip/hip_runtime.h>
#include <hip/hip_bf16.h>
#include <math.h>

// ---------------------------------------------------------------------------
// Problem geometry (fixed by the reference): B=2, S=4096, H=4096, P=1024
//   M = B*S = 8192 rows, K = H = 4096, N = P = 1024
// Pipeline per projection (vision, then text, same scratch):
//   1) ln_bf16:   LayerNorm rows of x (fp32) -> bf16 A matrix in ws (64 MB)
//   2) wt_bf16:   W (K x N fp32) -> W^T (N x K bf16) in ws (8 MB)
//   3) gemm_bf16: C = A @ W + bias via v_wmma_f32_16x16x32_bf16
// Both scratch matrices fit in the 192 MB L2, so the GEMM's fragment re-reads
// are L2-resident; only x (fp32), W (fp32) and C (fp32) touch HBM once.
// ---------------------------------------------------------------------------

typedef __attribute__((ext_vector_type(8)))  __bf16 v8bf;
typedef __attribute__((ext_vector_type(16))) __bf16 v16bf;
typedef __attribute__((ext_vector_type(8)))  float  v8f;

static constexpr int GM = 8192;   // rows (B*S)
static constexpr int GK = 4096;   // H
static constexpr int GN = 1024;   // P

// ------------------------- Pass 1: LayerNorm -> bf16 -----------------------
// One block per row; 256 threads reduce 4096 elements.
__global__ void ln_bf16_kernel(const float* __restrict__ x,
                               const float* __restrict__ ln_scale,
                               const float* __restrict__ ln_bias,
                               __bf16* __restrict__ out)
{
    const int row = blockIdx.x;
    const int t   = threadIdx.x;                 // 0..255
    const float* xr = x + (size_t)row * GK;

    float sum = 0.f, sq = 0.f;
    #pragma unroll
    for (int i = 0; i < 16; ++i) {
        float v = xr[t + i * 256];
        sum += v;
        sq  += v * v;
    }

    __shared__ float s1[256], s2[256];
    s1[t] = sum; s2[t] = sq;
    __syncthreads();
    for (int off = 128; off > 0; off >>= 1) {
        if (t < off) { s1[t] += s1[t + off]; s2[t] += s2[t + off]; }
        __syncthreads();
    }
    const float mu  = s1[0] * (1.f / (float)GK);
    const float var = s2[0] * (1.f / (float)GK) - mu * mu;
    const float rs  = rsqrtf(var + 1e-6f);

    __bf16* orow = out + (size_t)row * GK;
    #pragma unroll
    for (int i = 0; i < 16; ++i) {
        int h = t + i * 256;
        float v = (xr[h] - mu) * rs * ln_scale[h] + ln_bias[h];
        orow[h] = (__bf16)v;
    }
}

// ---------------- Pass 2: W (K x N, fp32) -> W^T (N x K, bf16) -------------
// LDS-tiled transpose so both the fp32 reads and bf16 writes are coalesced.
__global__ void wt_bf16_kernel(const float* __restrict__ W,
                               __bf16* __restrict__ Wt)
{
    __shared__ __bf16 tile[32][33];
    const int n0 = blockIdx.x * 32;   // along N (1024)
    const int k0 = blockIdx.y * 32;   // along K (4096)
    const int tx = threadIdx.x;       // 0..31
    const int ty = threadIdx.y;       // 0..7

    #pragma unroll
    for (int j = 0; j < 32; j += 8)
        tile[ty + j][tx] = (__bf16)W[(size_t)(k0 + ty + j) * GN + (n0 + tx)];
    __syncthreads();
    #pragma unroll
    for (int j = 0; j < 32; j += 8)
        Wt[(size_t)(n0 + ty + j) * GK + (k0 + tx)] = tile[tx][ty + j];
}

// --------------------------- Pass 3: WMMA GEMM -----------------------------
// A fragment (16x32 bf16, ISA 7.12.2): lane(0..15) row M=lane holds
//   K = {c..c+7} in v0..v3 and {c+16..c+23} in v4..v7; lanes 16..31 start at c+8.
// => two contiguous 16B loads at (+0, +32B) from a row-major A row.
__device__ __forceinline__ v16bf load_frag_a(const __bf16* p)
{
    v8bf lo = *(const v8bf*)(p);
    v8bf hi = *(const v8bf*)(p + 16);
    return __builtin_shufflevector(lo, hi, 0,1,2,3,4,5,6,7,8,9,10,11,12,13,14,15);
}
// B fragment (32x16 bf16): lane(0..15) col N=lane holds K = {c..c+7} in
// v0..v3 and {c+16..c+23} in v4..v7; lanes 16..31 start at c+8.
// => two contiguous 16B loads at (+0, +32B) from a row of W^T (N x K).
__device__ __forceinline__ v16bf load_frag_b(const __bf16* p)
{
    v8bf lo = *(const v8bf*)(p);
    v8bf hi = *(const v8bf*)(p + 16);
    return __builtin_shufflevector(lo, hi, 0,1,2,3,4,5,6,7,8,9,10,11,12,13,14,15);
}

// Block: 256 threads = 8 waves, arranged 4(M) x 2(N).
// Wave tile: 32(M) x 64(N) = 2x4 WMMA tiles -> 12 b128 loads feed 8 WMMAs.
// Block tile: 128(M) x 128(N). Grid: (GN/128, GM/128) = (8, 64).
// K loop: 4096/32 = 128 steps.
__global__ void gemm_bf16_kernel(const __bf16* __restrict__ A,   // M x K
                                 const __bf16* __restrict__ Bt,  // N x K
                                 const float* __restrict__ bias, // N
                                 float* __restrict__ C)          // M x N
{
    const int lane = threadIdx.x & 31;
    const int wave = threadIdx.x >> 5;
    const int l15  = lane & 15;
    const int lhi  = lane >> 4;

    const int m0 = blockIdx.y * 128 + (wave & 3) * 32;
    const int n0 = blockIdx.x * 128 + (wave >> 2) * 64;

    v8f acc[2][4] = {};

    const __bf16* ap[2];
    #pragma unroll
    for (int mi = 0; mi < 2; ++mi)
        ap[mi] = A + (size_t)(m0 + mi * 16 + l15) * GK + lhi * 8;

    const __bf16* bp[4];
    #pragma unroll
    for (int ni = 0; ni < 4; ++ni)
        bp[ni] = Bt + (size_t)(n0 + ni * 16 + l15) * GK + lhi * 8;

    for (int k = 0; k < GK; k += 32) {
        // prefetch the next K-step into cache (global_prefetch_b8)
        __builtin_prefetch(ap[0] + k + 32, 0, 1);
        __builtin_prefetch(bp[0] + k + 32, 0, 1);

        v16bf a[2];
        #pragma unroll
        for (int mi = 0; mi < 2; ++mi) a[mi] = load_frag_a(ap[mi] + k);

        v16bf b[4];
        #pragma unroll
        for (int ni = 0; ni < 4; ++ni) b[ni] = load_frag_b(bp[ni] + k);

        #pragma unroll
        for (int mi = 0; mi < 2; ++mi)
            #pragma unroll
            for (int ni = 0; ni < 4; ++ni)
                acc[mi][ni] = __builtin_amdgcn_wmma_f32_16x16x32_bf16(
                    false, a[mi], false, b[ni], (short)0, acc[mi][ni], false, false);
    }

    // C/D layout: lanes 0-15: VGPR r -> (M = r,  N = lane);
    //             lanes 16-31: VGPR r -> (M = 8+r, N = lane-16).
    const int row0 = m0 + lhi * 8;
    #pragma unroll
    for (int ni = 0; ni < 4; ++ni) {
        const int col = n0 + ni * 16 + l15;
        const float bb = bias[col];
        #pragma unroll
        for (int mi = 0; mi < 2; ++mi) {
            const int row = row0 + mi * 16;
            #pragma unroll
            for (int r = 0; r < 8; ++r)
                C[(size_t)(row + r) * GN + col] = acc[mi][ni][r] + bb;
        }
    }
}

// ---------------------------------------------------------------------------
extern "C" void kernel_launch(void* const* d_in, const int* in_sizes, int n_in,
                              void* d_out, int out_size, void* d_ws, size_t ws_size,
                              hipStream_t stream)
{
    (void)in_sizes; (void)n_in; (void)out_size; (void)ws_size;

    const float* vision    = (const float*)d_in[0];
    const float* text      = (const float*)d_in[1];
    const float* v_ln_s    = (const float*)d_in[2];
    const float* v_ln_b    = (const float*)d_in[3];
    const float* v_kernel  = (const float*)d_in[4];
    const float* v_bias    = (const float*)d_in[5];
    const float* t_ln_s    = (const float*)d_in[6];
    const float* t_ln_b    = (const float*)d_in[7];
    const float* t_kernel  = (const float*)d_in[8];
    const float* t_bias    = (const float*)d_in[9];

    // scratch: A (8192x4096 bf16 = 64 MB) + W^T (1024x4096 bf16 = 8 MB),
    // reused sequentially for vision then text -> both stay L2-resident.
    __bf16* Abuf  = (__bf16*)d_ws;
    __bf16* Wtbuf = Abuf + (size_t)GM * GK;

    float* out_v = (float*)d_out;
    float* out_t = out_v + (size_t)GM * GN;

    const dim3 ln_grid(GM), ln_block(256);
    const dim3 wt_grid(GN / 32, GK / 32), wt_block(32, 8);
    const dim3 gm_grid(GN / 128, GM / 128), gm_block(256);

    // vision projection
    ln_bf16_kernel<<<ln_grid, ln_block, 0, stream>>>(vision, v_ln_s, v_ln_b, Abuf);
    wt_bf16_kernel<<<wt_grid, wt_block, 0, stream>>>(v_kernel, Wtbuf);
    gemm_bf16_kernel<<<gm_grid, gm_block, 0, stream>>>(Abuf, Wtbuf, v_bias, out_v);

    // text projection (stream-ordered reuse of the same scratch)
    ln_bf16_kernel<<<ln_grid, ln_block, 0, stream>>>(text, t_ln_s, t_ln_b, Abuf);
    wt_bf16_kernel<<<wt_grid, wt_block, 0, stream>>>(t_kernel, Wtbuf);
    gemm_bf16_kernel<<<gm_grid, gm_block, 0, stream>>>(Abuf, Wtbuf, t_bias, out_t);
}
